// MultiHeadAttention_88433376624796
// MI455X (gfx1250) — matrix operand order repaired and checked by reading, not compiled, not run on hardware
//
#include <hip/hip_runtime.h>

// Problem constants (match reference)
#define BB 2
#define SS 2048
#define DD 1024
#define HH 16
#define HDIM 64

typedef __attribute__((ext_vector_type(16))) _Float16 v16h;
typedef __attribute__((ext_vector_type(8)))  _Float16 v8h;
typedef __attribute__((ext_vector_type(8)))  float    v8f;

// ---------------------------------------------------------------------------
// WMMA helpers (gfx1250 / CDNA5, wave32, v_wmma_f32_16x16x32_f16)
// ---------------------------------------------------------------------------

static __device__ __forceinline__ v8f wmma_f16(v16h a, v16h b, v8f c) {
  // 8 args: (neg_a, A, neg_b, B, c_mod, C, reuse_a, reuse_b)
  return __builtin_amdgcn_wmma_f32_16x16x32_f16(
      false, a, false, b, (short)0, c, false, false);
}

// Scheduling fence: nothing may be reordered across it.
static __device__ __forceinline__ void sched_fence() {
  __builtin_amdgcn_sched_barrier(0);
}

// A-matrix fragment (16x32, f16): lane L holds row m=L&15,
// halves cover K = 16*(L>=16) + {0..15}.
static __device__ __forceinline__ v16h load_frag_a(const _Float16* rowp, int hi) {
  return *(const v16h*)(rowp + (hi ? 16 : 0));
}

// B-matrix fragment (32x16, f16), operand stored as B^T row-major (N x K):
// lane L holds column n=L&15; halves cover K = 16*hi + {0..15}.
static __device__ __forceinline__ v16h load_frag_b(const _Float16* rowp, int hi) {
  return *(const v16h*)(rowp + (hi ? 16 : 0));
}

// ---------------------------------------------------------------------------
// Pass 0: fp32 -> fp16 conversion
// ---------------------------------------------------------------------------
__global__ void cvt_f32_f16(const float* __restrict__ src,
                            _Float16* __restrict__ dst, int n) {
  int i = blockIdx.x * blockDim.x + threadIdx.x;
  if (i < n) dst[i] = (_Float16)src[i];
}

// ---------------------------------------------------------------------------
// Pass 1: per-head projections.  grid = (S/32, B*H, 3), block = 32 (1 wave).
// Each wave: 32 rows x 64 cols; ping-pong double buffer with sched fences so
// refill loads land directly in dead operand registers (no copies/nops).
// z=0: Q -> [B,H,S,64];  z=1: K -> [B,H,S,64];  z=2: V -> Vt [B,H,64,S].
// ---------------------------------------------------------------------------
__global__ __launch_bounds__(32) void proj_kernel(
    const _Float16* __restrict__ Xq, const _Float16* __restrict__ Xk,
    const _Float16* __restrict__ Xv, const _Float16* __restrict__ Wq,
    const _Float16* __restrict__ Wk, const _Float16* __restrict__ Wv,
    const float* __restrict__ bq, const float* __restrict__ bk,
    const float* __restrict__ bv, _Float16* __restrict__ Qo,
    _Float16* __restrict__ Ko, _Float16* __restrict__ Vt) {
  const int lane = threadIdx.x;
  const int nn = lane & 15;
  const int hi = lane >> 4;
  const int s0 = blockIdx.x * 32;
  const int bh = blockIdx.y;            // b*H + h
  const int h  = bh & (HH - 1);
  const int b  = bh >> 4;
  const int pz = blockIdx.z;

  const _Float16* X    = (pz == 0) ? Xq : (pz == 1) ? Xk : Xv;
  const _Float16* W    = (pz == 0) ? Wq : (pz == 1) ? Wk : Wv;
  const float*    bias = (pz == 0) ? bq : (pz == 1) ? bk : bv;

  const v8f vzero = {0.f, 0.f, 0.f, 0.f, 0.f, 0.f, 0.f, 0.f};
  v8f acc0[4] = {vzero, vzero, vzero, vzero};
  v8f acc1[4] = {vzero, vzero, vzero, vzero};

  const _Float16* arow0 = X + (size_t)(b * SS + s0 + nn) * DD;   // rows s0..+15
  const _Float16* arow1 = arow0 + (size_t)16 * DD;               // rows +16..+31
  const _Float16* brow[4];
#pragma unroll
  for (int j = 0; j < 4; ++j)
    brow[j] = W + (size_t)(h * HDIM + j * 16 + nn) * DD;         // B^T row n = nn

  // ping-pong buffers A (even k-steps) and B (odd k-steps)
  v16h a0A = load_frag_a(arow0, hi);
  v16h a1A = load_frag_a(arow1, hi);
  v16h bfA[4];
#pragma unroll
  for (int j = 0; j < 4; ++j) bfA[j] = load_frag_b(brow[j], hi);
  v16h a0B = load_frag_a(arow0 + 32, hi);
  v16h a1B = load_frag_a(arow1 + 32, hi);
  v16h bfB[4];
#pragma unroll
  for (int j = 0; j < 4; ++j) bfB[j] = load_frag_b(brow[j] + 32, hi);

  int k0 = 64;
  for (int it = 0; it < DD / 64 - 1; ++it) {
#pragma unroll
    for (int j = 0; j < 4; ++j) {
      acc0[j] = wmma_f16(a0A, bfA[j], acc0[j]);
      acc1[j] = wmma_f16(a1A, bfA[j], acc1[j]);
    }
    sched_fence();                       // buf A now dead: refill in place
    a0A = load_frag_a(arow0 + k0, hi);
    a1A = load_frag_a(arow1 + k0, hi);
#pragma unroll
    for (int j = 0; j < 4; ++j) bfA[j] = load_frag_b(brow[j] + k0, hi);
    sched_fence();                       // A-loads in flight during B WMMAs
#pragma unroll
    for (int j = 0; j < 4; ++j) {
      acc0[j] = wmma_f16(a0B, bfB[j], acc0[j]);
      acc1[j] = wmma_f16(a1B, bfB[j], acc1[j]);
    }
    sched_fence();                       // buf B now dead: refill in place
    a0B = load_frag_a(arow0 + k0 + 32, hi);
    a1B = load_frag_a(arow1 + k0 + 32, hi);
#pragma unroll
    for (int j = 0; j < 4; ++j) bfB[j] = load_frag_b(brow[j] + k0 + 32, hi);
    sched_fence();
    k0 += 64;
  }
#pragma unroll
  for (int j = 0; j < 4; ++j) {
    acc0[j] = wmma_f16(a0A, bfA[j], acc0[j]);
    acc1[j] = wmma_f16(a1A, bfA[j], acc1[j]);
  }
#pragma unroll
  for (int j = 0; j < 4; ++j) {
    acc0[j] = wmma_f16(a0B, bfB[j], acc0[j]);
    acc1[j] = wmma_f16(a1B, bfB[j], acc1[j]);
  }

  float bb[4];
#pragma unroll
  for (int j = 0; j < 4; ++j) bb[j] = bias[h * HDIM + j * 16 + nn];

  if (pz < 2) {
    _Float16* dst = (pz == 0) ? Qo : Ko;   // [B,H,S,64]
#pragma unroll
    for (int j = 0; j < 4; ++j)
#pragma unroll
      for (int r = 0; r < 8; ++r) {
        int s = s0 + r + 8 * hi;           // C/D row mapping
        dst[((size_t)bh * SS + s) * HDIM + j * 16 + nn] =
            (_Float16)(acc0[j][r] + bb[j]);
        dst[((size_t)bh * SS + s + 16) * HDIM + j * 16 + nn] =
            (_Float16)(acc1[j][r] + bb[j]);
      }
  } else {                                  // V transposed: Vt[B,H,64,S]
#pragma unroll
    for (int j = 0; j < 4; ++j) {
      v8h pv0, pv1;
#pragma unroll
      for (int r = 0; r < 8; ++r) {
        pv0[r] = (_Float16)(acc0[j][r] + bb[j]);
        pv1[r] = (_Float16)(acc1[j][r] + bb[j]);
      }
      _Float16* vr = Vt + ((size_t)bh * HDIM + j * 16 + nn) * SS + s0 + 8 * hi;
      *(v8h*)(vr)      = pv0;
      *(v8h*)(vr + 16) = pv1;
    }
  }
}

// ---------------------------------------------------------------------------
// Pass 2: flash attention.  grid = (S/16, B*H), block = 32 (1 wave).
// Each wave: 16 query rows, online softmax over 32-key blocks.
// K-fragments prefetched one block ahead; V-fragments issued before score
// WMMAs so the softmax VALU tail hides their latency.  unroll 2 lets the
// compiler ping-pong the kf registers (no loop-carried copies).
// ---------------------------------------------------------------------------
__global__ __launch_bounds__(32) void flash_kernel(
    const _Float16* __restrict__ Q, const _Float16* __restrict__ K,
    const _Float16* __restrict__ Vt, _Float16* __restrict__ AO) {
  __shared__ __align__(32) _Float16 lds[16 * 32];  // P tile, D-layout -> A-layout

  const int lane = threadIdx.x;
  const int nn = lane & 15;
  const int hi = lane >> 4;
  const int q0 = blockIdx.x * 16;
  const int bh = blockIdx.y;
  const int h = bh & (HH - 1);
  const int b = bh >> 4;

  const _Float16* Qb = Q + (size_t)bh * SS * HDIM;
  const _Float16* Kb = K + (size_t)bh * SS * HDIM;
  const _Float16* Vb = Vt + (size_t)bh * HDIM * SS;

  const _Float16* qrow = Qb + (size_t)(q0 + nn) * HDIM;
  v16h qa0 = load_frag_a(qrow, hi);        // K-dim 0..31
  v16h qa1 = load_frag_a(qrow + 32, hi);   // K-dim 32..63

  const v8f vzero = {0.f, 0.f, 0.f, 0.f, 0.f, 0.f, 0.f, 0.f};
  v8f O[4] = {vzero, vzero, vzero, vzero};
  float l[8], mr[8];
#pragma unroll
  for (int r = 0; r < 8; ++r) { l[r] = 0.f; mr[r] = -1e30f; }

  // prefetch K fragments for the first key block
  // kf[0],kf[1]: keys t0..t0+15 (k-slices 0..31 / 32..63)
  // kf[2],kf[3]: keys t0+16..t0+31
  v16h kf[4];
  {
    const _Float16* kr0 = Kb + (size_t)nn * HDIM;
    const _Float16* kr1 = kr0 + 16 * HDIM;
    kf[0] = load_frag_b(kr0, hi);
    kf[1] = load_frag_b(kr0 + 32, hi);
    kf[2] = load_frag_b(kr1, hi);
    kf[3] = load_frag_b(kr1 + 32, hi);
  }

#pragma unroll 2
  for (int t0 = 0; t0 < SS; t0 += 32) {
    // issue V-fragment loads for this block early
    v16h vf[4];
#pragma unroll
    for (int j = 0; j < 4; ++j)
      vf[j] = load_frag_b(Vb + (size_t)(j * 16 + nn) * SS + t0, hi);

    // scores: 2 tiles of 16x16 (keys t0..+15 and t0+16..+31)
    v8f sa = vzero, sb = vzero;
    sa = wmma_f16(qa0, kf[0], sa);
    sa = wmma_f16(qa1, kf[1], sa);
    sb = wmma_f16(qa0, kf[2], sb);
    sb = wmma_f16(qa1, kf[3], sb);

    // prefetch K fragments for next block (wraps harmlessly on last iter)
    {
      int t1 = (t0 + 32 < SS) ? (t0 + 32) : 0;
      const _Float16* kr0 = Kb + (size_t)(t1 + nn) * HDIM;
      const _Float16* kr1 = kr0 + 16 * HDIM;
      kf[0] = load_frag_b(kr0, hi);
      kf[1] = load_frag_b(kr0 + 32, hi);
      kf[2] = load_frag_b(kr1, hi);
      kf[3] = load_frag_b(kr1 + 32, hi);
    }

    // online softmax; lane covers rows r + 8*hi, cols nn / 16+nn
#pragma unroll
    for (int r = 0; r < 8; ++r) {
      float a = sa[r] * 0.125f;            // 1/sqrt(64)
      float c = sb[r] * 0.125f;
      float t = fmaxf(a, c);
      t = fmaxf(t, __shfl_xor(t, 1));
      t = fmaxf(t, __shfl_xor(t, 2));
      t = fmaxf(t, __shfl_xor(t, 4));
      t = fmaxf(t, __shfl_xor(t, 8));      // max over 16-lane half (one row)
      float mn = fmaxf(mr[r], t);
      float e0 = __expf(a - mn);
      float e1 = __expf(c - mn);
      float rs = e0 + e1;
      rs += __shfl_xor(rs, 1);
      rs += __shfl_xor(rs, 2);
      rs += __shfl_xor(rs, 4);
      rs += __shfl_xor(rs, 8);             // row sum
      float alpha = __expf(mr[r] - mn);
      l[r] = l[r] * alpha + rs;
      mr[r] = mn;
      int row = r + 8 * hi;
      lds[row * 32 + nn]      = (_Float16)e0;
      lds[row * 32 + 16 + nn] = (_Float16)e1;
      O[0][r] *= alpha;
      O[1][r] *= alpha;
      O[2][r] *= alpha;
      O[3][r] *= alpha;
    }
    __syncthreads();                        // single-wave WG: near-free
    v16h pa = load_frag_a(lds + nn * 32, hi);  // P as A-fragment (16x32)
    __syncthreads();

    // O += P @ V_block   (B^T rows = Vt rows, K = key-local 0..31)
#pragma unroll
    for (int j = 0; j < 4; ++j) O[j] = wmma_f16(pa, vf[j], O[j]);
  }

  // normalize and write head-concat output [B,S,D] (fp16)
#pragma unroll
  for (int r = 0; r < 8; ++r) {
    float inv = 1.0f / l[r];
    int row = q0 + r + 8 * hi;
#pragma unroll
    for (int j = 0; j < 4; ++j)
      AO[((size_t)b * SS + row) * DD + h * HDIM + j * 16 + nn] =
          (_Float16)(O[j][r] * inv);
  }
}

// ---------------------------------------------------------------------------
// Pass 3: output projection  out = AO @ Wo^T + bo   (fp32 out)
// grid = ((B*S)/32, D/64), block = 32 (1 wave), 32x64 tile per wave,
// ping-pong double-buffered k-loop with sched fences, 8 WMMA / k-step.
// ---------------------------------------------------------------------------
__global__ __launch_bounds__(32) void outproj_kernel(
    const _Float16* __restrict__ AO, const _Float16* __restrict__ Wo,
    const float* __restrict__ bo, float* __restrict__ out) {
  const int lane = threadIdx.x;
  const int nn = lane & 15;
  const int hi = lane >> 4;
  const int m0 = blockIdx.x * 32;
  const int n0 = blockIdx.y * 64;

  const v8f vzero = {0.f, 0.f, 0.f, 0.f, 0.f, 0.f, 0.f, 0.f};
  v8f acc0[4] = {vzero, vzero, vzero, vzero};
  v8f acc1[4] = {vzero, vzero, vzero, vzero};

  const _Float16* arow0 = AO + (size_t)(m0 + nn) * DD;
  const _Float16* arow1 = arow0 + (size_t)16 * DD;
  const _Float16* brow[4];
#pragma unroll
  for (int j = 0; j < 4; ++j)
    brow[j] = Wo + (size_t)(n0 + j * 16 + nn) * DD;   // Wo [out,in] = B^T rows

  v16h a0A = load_frag_a(arow0, hi);
  v16h a1A = load_frag_a(arow1, hi);
  v16h bfA[4];
#pragma unroll
  for (int j = 0; j < 4; ++j) bfA[j] = load_frag_b(brow[j], hi);
  v16h a0B = load_frag_a(arow0 + 32, hi);
  v16h a1B = load_frag_a(arow1 + 32, hi);
  v16h bfB[4];
#pragma unroll
  for (int j = 0; j < 4; ++j) bfB[j] = load_frag_b(brow[j] + 32, hi);

  int k0 = 64;
  for (int it = 0; it < DD / 64 - 1; ++it) {
#pragma unroll
    for (int j = 0; j < 4; ++j) {
      acc0[j] = wmma_f16(a0A, bfA[j], acc0[j]);
      acc1[j] = wmma_f16(a1A, bfA[j], acc1[j]);
    }
    sched_fence();
    a0A = load_frag_a(arow0 + k0, hi);
    a1A = load_frag_a(arow1 + k0, hi);
#pragma unroll
    for (int j = 0; j < 4; ++j) bfA[j] = load_frag_b(brow[j] + k0, hi);
    sched_fence();
#pragma unroll
    for (int j = 0; j < 4; ++j) {
      acc0[j] = wmma_f16(a0B, bfB[j], acc0[j]);
      acc1[j] = wmma_f16(a1B, bfB[j], acc1[j]);
    }
    sched_fence();
    a0B = load_frag_a(arow0 + k0 + 32, hi);
    a1B = load_frag_a(arow1 + k0 + 32, hi);
#pragma unroll
    for (int j = 0; j < 4; ++j) bfB[j] = load_frag_b(brow[j] + k0 + 32, hi);
    sched_fence();
    k0 += 64;
  }
#pragma unroll
  for (int j = 0; j < 4; ++j) {
    acc0[j] = wmma_f16(a0A, bfA[j], acc0[j]);
    acc1[j] = wmma_f16(a1A, bfA[j], acc1[j]);
  }
#pragma unroll
  for (int j = 0; j < 4; ++j) {
    acc0[j] = wmma_f16(a0B, bfB[j], acc0[j]);
    acc1[j] = wmma_f16(a1B, bfB[j], acc1[j]);
  }

#pragma unroll
  for (int j = 0; j < 4; ++j) {
    float bb = bo[n0 + j * 16 + nn];
#pragma unroll
    for (int r = 0; r < 8; ++r) {
      out[(size_t)(m0 + r + 8 * hi) * DD + n0 + j * 16 + nn] = acc0[j][r] + bb;
      out[(size_t)(m0 + 16 + r + 8 * hi) * DD + n0 + j * 16 + nn] =
          acc1[j][r] + bb;
    }
  }
}

// ---------------------------------------------------------------------------
// Host-side launcher
// ---------------------------------------------------------------------------
extern "C" void kernel_launch(void* const* d_in, const int* in_sizes, int n_in,
                              void* d_out, int out_size, void* d_ws,
                              size_t ws_size, hipStream_t stream) {
  (void)in_sizes; (void)n_in; (void)out_size; (void)ws_size;

  const float* q  = (const float*)d_in[0];
  const float* k  = (const float*)d_in[1];
  const float* v  = (const float*)d_in[2];
  const float* Wq = (const float*)d_in[3];
  const float* bq = (const float*)d_in[4];
  const float* Wk = (const float*)d_in[5];
  const float* bk = (const float*)d_in[6];
  const float* Wv = (const float*)d_in[7];
  const float* bv = (const float*)d_in[8];
  const float* Wo = (const float*)d_in[9];
  const float* bo = (const float*)d_in[10];
  float* out = (float*)d_out;

  const size_t NX  = (size_t)BB * SS * DD;      // 4,194,304
  const size_t NW  = (size_t)HH * HDIM * DD;    // 1,048,576
  const size_t NWO = (size_t)DD * DD;           // 1,048,576

  // fp16 workspace layout (total 33,554,432 halves = 64 MiB)
  _Float16* ws  = (_Float16*)d_ws;
  _Float16* Xq  = ws;
  _Float16* Xk  = Xq + NX;
  _Float16* Xv  = Xk + NX;
  _Float16* Wqh = Xv + NX;
  _Float16* Wkh = Wqh + NW;
  _Float16* Wvh = Wkh + NW;
  _Float16* Woh = Wvh + NW;
  _Float16* Qh  = Woh + NWO;
  _Float16* Kh  = Qh + NX;
  _Float16* Vth = Kh + NX;
  _Float16* AO  = Vth + NX;

  const int CB = 256;
  cvt_f32_f16<<<(int)((NX + CB - 1) / CB), CB, 0, stream>>>(q, Xq, (int)NX);
  cvt_f32_f16<<<(int)((NX + CB - 1) / CB), CB, 0, stream>>>(k, Xk, (int)NX);
  cvt_f32_f16<<<(int)((NX + CB - 1) / CB), CB, 0, stream>>>(v, Xv, (int)NX);
  cvt_f32_f16<<<(int)((NW + CB - 1) / CB), CB, 0, stream>>>(Wq, Wqh, (int)NW);
  cvt_f32_f16<<<(int)((NW + CB - 1) / CB), CB, 0, stream>>>(Wk, Wkh, (int)NW);
  cvt_f32_f16<<<(int)((NW + CB - 1) / CB), CB, 0, stream>>>(Wv, Wvh, (int)NW);
  cvt_f32_f16<<<(int)((NWO + CB - 1) / CB), CB, 0, stream>>>(Wo, Woh, (int)NWO);

  proj_kernel<<<dim3(SS / 32, BB * HH, 3), 32, 0, stream>>>(
      Xq, Xk, Xv, Wqh, Wkh, Wvh, bq, bk, bv, Qh, Kh, Vth);

  flash_kernel<<<dim3(SS / 16, BB * HH), 32, 0, stream>>>(Qh, Kh, Vth, AO);

  outproj_kernel<<<dim3((BB * SS) / 32, DD / 64), 32, 0, stream>>>(
      AO, Woh, bo, out);
}
